// MHPLSTMCore_17454747090987
// MI455X (gfx1250) — matrix operand order repaired and checked by reading, not compiled, not run on hardware
//
#include <hip/hip_runtime.h>
#include <hip/hip_bf16.h>
#include <math.h>

typedef __attribute__((ext_vector_type(16))) _Float16 v16h;
typedef __attribute__((ext_vector_type(8)))  _Float16 v8h;
typedef __attribute__((ext_vector_type(8)))  float    v8f;

#define Bn   4
#define Sn   2048
#define Hn   8
#define Dn   128
#define FHn  512
#define TOK  (Bn * Sn)          // 8192 tokens

__device__ __forceinline__ float sigmoidf_(float x) { return 1.0f / (1.0f + expf(-x)); }

// Block-wide mean/rsqrt(var) over (256 threads x partials). wave32 shuffles + LDS.
__device__ __forceinline__ float2 block_stats_256(float s, float ss, float invN) {
    __shared__ float red0[8], red1[8];
    const int lane = threadIdx.x & 31;
    const int w    = threadIdx.x >> 5;
#pragma unroll
    for (int o = 16; o > 0; o >>= 1) {
        s  += __shfl_down(s,  o, 32);
        ss += __shfl_down(ss, o, 32);
    }
    if (lane == 0) { red0[w] = s; red1[w] = ss; }
    __syncthreads();
    if (w == 0) {
        float a = (lane < 8) ? red0[lane] : 0.0f;
        float b = (lane < 8) ? red1[lane] : 0.0f;
#pragma unroll
        for (int o = 4; o > 0; o >>= 1) {
            a += __shfl_down(a, o, 32);
            b += __shfl_down(b, o, 32);
        }
        if (lane == 0) { red0[0] = a; red1[0] = b; }
    }
    __syncthreads();
    float mean = red0[0] * invN;
    float var  = red1[0] * invN - mean * mean;
    return make_float2(mean, rsqrtf(var + 1e-6f));
}

// ---- f32 -> f16 weight conversion WITH transpose: W[h,k,n] -> Wt[h,n,k] -----
__global__ void mhp_cvt_wt(const float* __restrict__ src, _Float16* __restrict__ dst,
                           int K, int N) {
    int i = blockIdx.x * blockDim.x + threadIdx.x;
    int total = Hn * K * N;
    if (i < total) {
        int h = i / (K * N);
        int r = i % (K * N);
        int k = r / N, n = r % N;
        dst[(size_t)h * N * K + (size_t)n * K + k] = (_Float16)src[i];
    }
}

// ---- cumsum of shifted input over S (one lane per (b,h,d)) ------------------
__global__ void mhp_cumsum(const float* __restrict__ x, float* __restrict__ csum) {
    const int bh = blockIdx.x;              // 0..31
    const int b  = bh / Hn, h = bh % Hn;
    const int d  = threadIdx.x;             // 0..127
    float c = 0.0f;
    size_t base = ((size_t)b * Sn) * (Hn * Dn) + (size_t)h * Dn + d;
    for (int s = 0; s < Sn; ++s) {
        size_t idx = base + (size_t)s * (Hn * Dn);
        csum[idx] = c;                      // cumsum of shifted = prefix-exclusive sum
        c += x[idx];
    }
}

// ---- LN(csum) over (H,D)=1024 per token; emit gh = [x | ln(csum)] in f16 ----
__global__ __launch_bounds__(256) void mhp_ghbuild(
    const float* __restrict__ x, const float* __restrict__ csum,
    const float* __restrict__ g, const float* __restrict__ bb,
    _Float16* __restrict__ gh)
{
    const size_t t = blockIdx.x;
    const int tid = threadIdx.x;
    float v[4]; float s = 0.f, ss = 0.f;
#pragma unroll
    for (int i = 0; i < 4; ++i) {
        int j = tid + i * 256;
        float val = csum[t * 1024 + j];
        v[i] = val; s += val; ss += val * val;
    }
    float2 st = block_stats_256(s, ss, 1.0f / 1024.0f);
#pragma unroll
    for (int i = 0; i < 4; ++i) {
        int j = tid + i * 256;
        int h = j >> 7, d = j & 127;
        float ln = (v[i] - st.x) * st.y * g[j] + bb[j];
        gh[t * 2048 + h * 256 + d]       = (_Float16)x[t * 1024 + j];
        gh[t * 2048 + h * 256 + 128 + d] = (_Float16)ln;
    }
}

// ---- Generic per-head batched GEMM via v_wmma_f32_16x16x32_f16 --------------
// C[t, h, n] = sum_k A[t*strideA + h*K + k] * Wt[h, n, k] + bias[h, n]
// Block: 256 threads = 8 waves. Block tile M=128, N=64, K-step 32, K unrolled.
// Each wave owns a 16(M) x 64(N) strip: one A fragment feeds 4 WMMAs / K-step.
// Double-buffered LDS (one barrier per K-step); both A and B tiles staged with
// 16B global_load_b128 -> ds_store_b128 (B pre-transposed to [n][k] in global).
template <int K, int N>
__global__ __launch_bounds__(256) void mhp_gemm_wmma(
    const _Float16* __restrict__ A, int strideA,
    const _Float16* __restrict__ Wt,     // (H, N, K) f16, pre-transposed
    const float* __restrict__ bias,
    float* __restrict__ C)
{
    constexpr int KT = 32;
    const int h    = blockIdx.z;
    const int m0   = blockIdx.x * 128;
    const int n0   = blockIdx.y * 64;
    const int tid  = threadIdx.x;
    const int wave = tid >> 5, lane = tid & 31;
    const int hi   = lane >> 4;       // half-wave group
    const int l15  = lane & 15;

    __shared__ _Float16 As[2][128][40];  // [m][k], 80B row stride (16B aligned)
    __shared__ _Float16 Bt[2][64][40];   // [n][k]

    const _Float16* Ah  = A  + (size_t)h * K;
    const _Float16* Wth = Wt + (size_t)h * N * K;

    // Stage one K-tile into LDS buffer `buf` (all 16B vector ops).
    auto stage = [&](int buf, int k0) {
        if (k0 + 2 * KT <= K - KT) {   // CDNA5 global_prefetch_b8, two tiles ahead
            __builtin_prefetch(Ah  + (size_t)(m0 + (tid >> 1)) * strideA + (k0 + 2 * KT), 0, 1);
            __builtin_prefetch(Wth + (size_t)(n0 + (tid >> 2)) * K       + (k0 + 2 * KT), 0, 1);
        }
        // A tile: 128 rows x 32 halves = 512 v8h chunks, 2 per thread
#pragma unroll
        for (int c = 0; c < 2; ++c) {
            int i = tid + c * 256;
            int r = i >> 2, col = (i & 3) * 8;
            *(v8h*)&As[buf][r][col] =
                *(const v8h*)&Ah[(size_t)(m0 + r) * strideA + (k0 + col)];
        }
        // B tile: 64 rows x 32 halves = 256 v8h chunks, 1 per thread
        {
            int n = tid >> 2, col = (tid & 3) * 8;
            *(v8h*)&Bt[buf][n][col] =
                *(const v8h*)&Wth[(size_t)(n0 + n) * K + (k0 + col)];
        }
    };

    v8f acc[4] = {};
    stage(0, 0);
    __syncthreads();

#pragma unroll
    for (int kk = 0; kk < K / KT; ++kk) {
        const int buf = kk & 1;
        if (kk + 1 < K / KT) stage(buf ^ 1, (kk + 1) * KT);

        // A fragment per CDNA5 16-bit A(16x32) striping:
        //   lane<16: K 0-7 | 16-23 ; lane>=16: K 8-15 | 24-31  (two 16B chunks)
        const _Float16* arow = &As[buf][wave * 16 + l15][0];
        v8h a0 = *(const v8h*)(arow + hi * 8);
        v8h a1 = *(const v8h*)(arow + 16 + hi * 8);
        v16h af = __builtin_shufflevector(a0, a1, 0, 1, 2, 3, 4, 5, 6, 7,
                                                  8, 9, 10, 11, 12, 13, 14, 15);
#pragma unroll
        for (int s = 0; s < 4; ++s) {
            // B fragment per B(32x16) striping: lane%16 = N col;
            //   lane<16: K 0-7 | 16-23 ; lane>=16: K 8-15 | 24-31 (two 16B chunks)
            const _Float16* brow = &Bt[buf][s * 16 + l15][hi * 8];
            v8h b0 = *(const v8h*)(brow);
            v8h b1 = *(const v8h*)(brow + 16);
            v16h bf = __builtin_shufflevector(b0, b1, 0, 1, 2, 3, 4, 5, 6, 7,
                                                      8, 9, 10, 11, 12, 13, 14, 15);
            acc[s] = __builtin_amdgcn_wmma_f32_16x16x32_f16(false, af, false, bf,
                                                            (short)0, acc[s], false, false);
        }
        __syncthreads();
    }

    // C/D layout: VGPR r -> M = r + 8*(lane/16); N = lane%16
#pragma unroll
    for (int s = 0; s < 4; ++s) {
        const int cn = n0 + s * 16 + l15;
        const float bv = bias[h * N + cn];
#pragma unroll
        for (int r = 0; r < 8; ++r) {
            int cm = m0 + wave * 16 + hi * 8 + r;
            C[(size_t)cm * ((size_t)Hn * N) + (size_t)h * N + cn] = acc[s][r] + bv;
        }
    }
}

// ---- LN over (H,2,D)=2048 per token -> sigmoid gates ------------------------
__global__ __launch_bounds__(256) void mhp_ifgln(
    const float* __restrict__ raw, const float* __restrict__ g, const float* __restrict__ bb,
    float* __restrict__ isig, float* __restrict__ fsig)
{
    const size_t t = blockIdx.x;
    const int tid = threadIdx.x;
    float v[8]; float s = 0.f, ss = 0.f;
#pragma unroll
    for (int i = 0; i < 8; ++i) {
        int j = tid + i * 256;
        float val = raw[t * 2048 + j];
        v[i] = val; s += val; ss += val * val;
    }
    float2 st = block_stats_256(s, ss, 1.0f / 2048.0f);
#pragma unroll
    for (int i = 0; i < 8; ++i) {
        int j = tid + i * 256;
        float y = (v[i] - st.x) * st.y * g[j] + bb[j];
        float sg = sigmoidf_(y);
        int h = j >> 8;
        int r = (j >> 7) & 1;
        int d = j & 127;
        size_t o = t * 1024 + h * 128 + d;
        if (r == 0) isig[o] = sg; else fsig[o] = sg;
    }
}

// ---- LN over (H,FH)=4096 per token + relu -> f16 ----------------------------
__global__ __launch_bounds__(256) void mhp_hidln(
    const float* __restrict__ raw, const float* __restrict__ g, const float* __restrict__ bb,
    _Float16* __restrict__ hidh)
{
    const size_t t = blockIdx.x;
    const int tid = threadIdx.x;
    float v[16]; float s = 0.f, ss = 0.f;
#pragma unroll
    for (int i = 0; i < 16; ++i) {
        int j = tid + i * 256;
        float val = raw[t * 4096 + j];
        v[i] = val; s += val; ss += val * val;
    }
    float2 st = block_stats_256(s, ss, 1.0f / 4096.0f);
#pragma unroll
    for (int i = 0; i < 16; ++i) {
        int j = tid + i * 256;
        float y = (v[i] - st.x) * st.y * g[j] + bb[j];
        y = y > 0.0f ? y : 0.0f;
        hidh[t * 4096 + j] = (_Float16)y;
    }
}

// ---- Sequential cell scan along S; also build ghc = [x | cell] in f16 -------
__global__ void mhp_scan(
    const float* __restrict__ x, const float* __restrict__ hidden,
    const float* __restrict__ fsig, const float* __restrict__ isig,
    const float* __restrict__ init_cx,
    float* __restrict__ cell, _Float16* __restrict__ ghc)
{
    const int bh = blockIdx.x;
    const int b  = bh / Hn, h = bh % Hn;
    const int d  = threadIdx.x;
    float c = init_cx[h * Dn + d];
    for (int s = 0; s < Sn; ++s) {
        size_t t   = (size_t)b * Sn + s;
        size_t idx = t * 1024 + h * 128 + d;
        c = fsig[idx] * c + isig[idx] * hidden[idx];
        cell[idx] = c;
        ghc[t * 2048 + h * 256 + d]       = (_Float16)x[idx];
        ghc[t * 2048 + h * 256 + 128 + d] = (_Float16)c;
    }
}

// ---- LN over (H,D)=1024 per token + sigmoid * cell -> output ----------------
__global__ __launch_bounds__(256) void mhp_ogln(
    const float* __restrict__ raw, const float* __restrict__ g, const float* __restrict__ bb,
    const float* __restrict__ cell, float* __restrict__ out)
{
    const size_t t = blockIdx.x;
    const int tid = threadIdx.x;
    float v[4]; float s = 0.f, ss = 0.f;
#pragma unroll
    for (int i = 0; i < 4; ++i) {
        int j = tid + i * 256;
        float val = raw[t * 1024 + j];
        v[i] = val; s += val; ss += val * val;
    }
    float2 st = block_stats_256(s, ss, 1.0f / 1024.0f);
#pragma unroll
    for (int i = 0; i < 4; ++i) {
        int j = tid + i * 256;
        float y = (v[i] - st.x) * st.y * g[j] + bb[j];
        out[t * 1024 + j] = sigmoidf_(y) * cell[t * 1024 + j];
    }
}

// -----------------------------------------------------------------------------
extern "C" void kernel_launch(void* const* d_in, const int* in_sizes, int n_in,
                              void* d_out, int out_size, void* d_ws, size_t ws_size,
                              hipStream_t stream)
{
    (void)in_sizes; (void)n_in; (void)out_size; (void)ws_size;

    const float* x       = (const float*)d_in[0];
    const float* W_ifg   = (const float*)d_in[1];
    const float* b_ifg   = (const float*)d_in[2];
    const float* W_hid1  = (const float*)d_in[3];
    const float* b_hid1  = (const float*)d_in[4];
    const float* hid_g   = (const float*)d_in[5];
    const float* hid_b   = (const float*)d_in[6];
    const float* W_hid2  = (const float*)d_in[7];
    const float* b_hid2  = (const float*)d_in[8];
    const float* W_og    = (const float*)d_in[9];
    const float* b_og    = (const float*)d_in[10];
    const float* og_g    = (const float*)d_in[11];
    const float* og_b    = (const float*)d_in[12];
    const float* csum_g  = (const float*)d_in[13];
    const float* csum_b  = (const float*)d_in[14];
    const float* ifg_g   = (const float*)d_in[15];
    const float* ifg_b   = (const float*)d_in[16];
    const float* init_cx = (const float*)d_in[17];
    float* out = (float*)d_out;
    char*  ws  = (char*)d_ws;

    constexpr size_t MB = 1ull << 20;
    // Workspace layout (buffers reused where lifetimes allow):
    _Float16* gh     = (_Float16*)(ws + 0);          // 32 MB: (TOK,H,256) f16
    _Float16* hidh   = (_Float16*)(ws + 32  * MB);   // 64 MB: (TOK,H,512) f16
    _Float16* ghc    = (_Float16*)(ws + 96  * MB);   // 32 MB: (TOK,H,256) f16
    float*    raw    = (float*)   (ws + 128 * MB);   // 128 MB: GEMM output scratch (reused 3x)
    float*    csum   = (float*)   (ws + 256 * MB);   // 32 MB: csum, later reused as `hidden`
    float*    hidden = csum;                          // alias (csum dead after ghbuild)
    float*    fsig   = (float*)   (ws + 288 * MB);   // 32 MB
    float*    isig   = (float*)   (ws + 320 * MB);   // 32 MB
    float*    cell   = (float*)   (ws + 352 * MB);   // 32 MB
    _Float16* wifg_t = (_Float16*)(ws + 384 * MB);   // 1 MB   (H, 256, 256) [n][k]
    _Float16* wh1_t  = (_Float16*)(ws + 385 * MB);   // 2 MB   (H, 512, 256)
    _Float16* wh2_t  = (_Float16*)(ws + 387 * MB);   // 1 MB   (H, 128, 512)
    _Float16* wog_t  = (_Float16*)(ws + 388 * MB);   // 0.5 MB (H, 128, 256)

    // Weight conversion f32 -> f16 with transpose to [h][n][k]
    mhp_cvt_wt<<<(Hn*256*256 + 255) / 256, 256, 0, stream>>>(W_ifg,  wifg_t, 256, 256);
    mhp_cvt_wt<<<(Hn*256*512 + 255) / 256, 256, 0, stream>>>(W_hid1, wh1_t,  256, 512);
    mhp_cvt_wt<<<(Hn*512*128 + 255) / 256, 256, 0, stream>>>(W_hid2, wh2_t,  512, 128);
    mhp_cvt_wt<<<(Hn*256*128 + 255) / 256, 256, 0, stream>>>(W_og,   wog_t,  256, 128);

    // 1) cumsum of shifted input
    mhp_cumsum<<<Bn * Hn, Dn, 0, stream>>>(x, csum);
    // 2) LN(csum) + build gh in f16
    mhp_ghbuild<<<TOK, 256, 0, stream>>>(x, csum, csum_g, csum_b, gh);
    // 3) ifg = gh @ W_ifg (K=256, N=256)
    mhp_gemm_wmma<256, 256><<<dim3(TOK / 128, 256 / 64, Hn), 256, 0, stream>>>(
        gh, 2048, wifg_t, b_ifg, raw);
    // 4) LN(ifg) over (H,2,D) -> sigmoid gates
    mhp_ifgln<<<TOK, 256, 0, stream>>>(raw, ifg_g, ifg_b, isig, fsig);
    // 5) hid = gh @ W_hid1 (K=256, N=512)
    mhp_gemm_wmma<256, 512><<<dim3(TOK / 128, 512 / 64, Hn), 256, 0, stream>>>(
        gh, 2048, wh1_t, b_hid1, raw);
    // 6) LN(hid) + relu -> f16
    mhp_hidln<<<TOK, 256, 0, stream>>>(raw, hid_g, hid_b, hidh);
    // 7) hidden = hid @ W_hid2 (K=512, N=128)
    mhp_gemm_wmma<512, 128><<<dim3(TOK / 128, 128 / 64, Hn), 256, 0, stream>>>(
        hidh, 4096, wh2_t, b_hid2, hidden);
    // 8) cell scan + build ghc
    mhp_scan<<<Bn * Hn, Dn, 0, stream>>>(x, hidden, fsig, isig, init_cx, cell, ghc);
    // 9) og = ghc @ W_og (K=256, N=128)
    mhp_gemm_wmma<256, 128><<<dim3(TOK / 128, 128 / 64, Hn), 256, 0, stream>>>(
        ghc, 2048, wog_t, b_og, raw);
    // 10) LN(og) + sigmoid * cell -> output
    mhp_ogln<<<TOK, 256, 0, stream>>>(raw, og_g, og_b, cell, out);
}
